// LlamaAttention_25769804331
// MI455X (gfx1250) — matrix operand order repaired and checked by reading, not compiled, not run on hardware
//
#include <hip/hip_runtime.h>
#include <hip/hip_bf16.h>

typedef unsigned short u16;
typedef __attribute__((ext_vector_type(16))) __bf16 v16bf;
typedef __attribute__((ext_vector_type(8)))  float  v8f;

union Frag16 { uint4 u[2]; v16bf v; };

static __device__ __forceinline__ u16 f2bf(float x) {
    union { float f; unsigned u; } v; v.f = x;
    unsigned r = v.u + 0x7FFFu + ((v.u >> 16) & 1u);
    return (u16)(r >> 16);
}

#define WMMA_BF16(A, B, C) \
    __builtin_amdgcn_wmma_f32_16x16x32_bf16(false, (A), false, (B), (short)0, (C), false, false)

// Async global->LDS copy (CDNA5 path, tracked by ASYNCcnt, no VGPR staging).
// lds = raw LDS byte offset (low 32 bits of the generic shared pointer),
// g   = 64-bit global address in a VGPR pair, GV addressing.
static __device__ __forceinline__ void async_ld128(unsigned lds, const u16* g) {
    asm volatile("global_load_async_to_lds_b128 %0, %1, off"
                 :: "v"(lds), "v"((unsigned long long)(size_t)g)
                 : "memory");
}

static __device__ __forceinline__ unsigned lds_off32(const void* p) {
    return (unsigned)(size_t)p;   // shared aperture: low 32 bits == LDS offset
}

// ---------------------------------------------------------------------------
// f32 -> bf16 conversion (8 elements / thread, vectorized)
// ---------------------------------------------------------------------------
__global__ __launch_bounds__(256) void cvt_f32_bf16(const float* __restrict__ src,
                                                    u16* __restrict__ dst, long n) {
    long i = ((long)blockIdx.x * 256 + threadIdx.x) * 8;
    if (i >= n) return;
    float4 f0 = *(const float4*)(src + i);
    float4 f1 = *(const float4*)(src + i + 4);
    uint4 o;
    o.x = (unsigned)f2bf(f0.x) | ((unsigned)f2bf(f0.y) << 16);
    o.y = (unsigned)f2bf(f0.z) | ((unsigned)f2bf(f0.w) << 16);
    o.z = (unsigned)f2bf(f1.x) | ((unsigned)f2bf(f1.y) << 16);
    o.w = (unsigned)f2bf(f1.z) | ((unsigned)f2bf(f1.w) << 16);
    *(uint4*)(dst + i) = o;
}

// ---------------------------------------------------------------------------
// bf16 TN GEMM: C[M,N](f32) = A[M,K](bf16, row-major) * B[N,K](bf16)^T
// 128x128 block tile, 8 waves, wave tile 32x64, K step 32.
// Double-buffered LDS filled by global_load_async_to_lds_b128; steady-state
// loop is branch-free (last iteration peeled), next tile's async copy is in
// flight while the current tile's 8 WMMAs execute.
// LDS pitch 40 u16 (80B) to dodge bank conflicts; 16B aligned for b128 ops.
// ---------------------------------------------------------------------------
__global__ __launch_bounds__(256) void gemm_bf16_tn(const u16* __restrict__ A,
                                                    const u16* __restrict__ B,
                                                    float* __restrict__ C,
                                                    int M, int N, int K) {
    __shared__ __align__(16) u16 As[2][128 * 40];
    __shared__ __align__(16) u16 Bs[2][128 * 40];

    const int bn = blockIdx.x, bm = blockIdx.y;
    const int tid  = threadIdx.x;
    const int w    = tid >> 5;
    const int lane = tid & 31;
    const int half = lane >> 4;
    const int ln   = lane & 15;
    const int wm = w & 3, wn = w >> 2;
    const int rm = wm * 32, rn = wn * 64;

    v8f acc[2][4];
    v8f zero = {0.f,0.f,0.f,0.f,0.f,0.f,0.f,0.f};
    for (int mt = 0; mt < 2; ++mt)
        for (int nt = 0; nt < 4; ++nt) acc[mt][nt] = zero;

    const int rowL = tid >> 1;              // 0..127: LDS row this thread fills
    const int seg  = (tid & 1) * 16;        // u16 offset within row (32B halves)
    const u16* aNext = A + (size_t)(bm * 128 + rowL) * K + seg;
    const u16* bNext = B + (size_t)(bn * 128 + rowL) * K + seg;
    unsigned aDst[2], bDst[2];
    aDst[0] = lds_off32(&As[0][rowL * 40 + seg]);
    aDst[1] = lds_off32(&As[1][rowL * 40 + seg]);
    bDst[0] = lds_off32(&Bs[0][rowL * 40 + seg]);
    bDst[1] = lds_off32(&Bs[1][rowL * 40 + seg]);

    // one K-tile's worth of fragment loads + WMMAs from LDS buffer 'cur'
    auto compute_tile = [&](int cur) {
        const u16* Ac = As[cur];
        const u16* Bc = Bs[cur];
        Frag16 af[2], bf[4];
        for (int mt = 0; mt < 2; ++mt) {
            const u16* p = Ac + (rm + mt * 16 + ln) * 40;
            af[mt].u[0] = *(const uint4*)(p + half * 8);        // K 0-7 / 8-15
            af[mt].u[1] = *(const uint4*)(p + 16 + half * 8);   // K 16-23 / 24-31
        }
        for (int nt = 0; nt < 4; ++nt) {
            const u16* p = Bc + (rn + nt * 16 + ln) * 40;
            bf[nt].u[0] = *(const uint4*)(p + half * 8);        // K 0-7 / 8-15
            bf[nt].u[1] = *(const uint4*)(p + 16 + half * 8);   // K 16-23 / 24-31
        }
        for (int mt = 0; mt < 2; ++mt)
            for (int nt = 0; nt < 4; ++nt)
                acc[mt][nt] = WMMA_BF16(af[mt].v, bf[nt].v, acc[mt][nt]);
    };

    const int niter = K >> 5;
    // prologue: fill buffer 0 (4 async b128 per thread)
    async_ld128(aDst[0],      aNext);
    async_ld128(aDst[0] + 32, aNext + 8);
    async_ld128(bDst[0],      bNext);
    async_ld128(bDst[0] + 32, bNext + 8);
    aNext += 32; bNext += 32;

    // steady state: branch-free, prefetch next tile while computing current
    for (int i = 0; i < niter - 1; ++i) {
        const int cur = i & 1;
        async_ld128(aDst[cur ^ 1],      aNext);
        async_ld128(aDst[cur ^ 1] + 32, aNext + 8);
        async_ld128(bDst[cur ^ 1],      bNext);
        async_ld128(bDst[cur ^ 1] + 32, bNext + 8);
        aNext += 32; bNext += 32;
        asm volatile("s_wait_asynccnt 0x4" ::: "memory");  // drain current tile
        __syncthreads();                    // current tile visible to all waves
        compute_tile(cur);
        __syncthreads();                    // reads done before next overwrite
    }

    // epilogue: last tile
    asm volatile("s_wait_asynccnt 0x0" ::: "memory");
    __syncthreads();
    compute_tile((niter - 1) & 1);

    for (int mt = 0; mt < 2; ++mt)
        for (int nt = 0; nt < 4; ++nt)
            for (int r = 0; r < 8; ++r) {
                int row = bm * 128 + rm + mt * 16 + r + half * 8;
                int col = bn * 128 + rn + nt * 16 + ln;
                C[(size_t)row * N + col] = acc[mt][nt][r];
            }
}

// ---------------------------------------------------------------------------
// RoPE + repack: qkv f32 [T,6144] -> Q bf16 [T,32,128] (pre-scaled by 1/sqrt(128)),
// K bf16 [T,8,128], V bf16 transposed [8][128][T] (for contiguous PV B-frags)
// ---------------------------------------------------------------------------
__global__ __launch_bounds__(256) void rope_pack(const float* __restrict__ qkv,
                                                 const int* __restrict__ pos,
                                                 u16* __restrict__ Qb,
                                                 u16* __restrict__ Kb,
                                                 u16* __restrict__ Vt) {
    const int t = blockIdx.x;
    const float* row = qkv + (size_t)t * 6144;
    const float p = (float)pos[t];
    const float qscale = 0.08838834764831845f;   // 128^-0.5
    const int tid = threadIdx.x;

    for (int i = tid; i < 2048; i += 256) {      // Q: 32 heads * 64 pairs
        int h = i >> 6, d = i & 63;
        float inv = __powf(10000.0f, -(float)d * (1.0f / 64.0f));
        float s, c; __sincosf(p * inv, &s, &c);
        float x1 = row[h * 128 + d], x2 = row[h * 128 + d + 64];
        Qb[(size_t)t * 4096 + h * 128 + d]      = f2bf((x1 * c - x2 * s) * qscale);
        Qb[(size_t)t * 4096 + h * 128 + d + 64] = f2bf((x2 * c + x1 * s) * qscale);
    }
    for (int i = tid; i < 512; i += 256) {       // K: 8 heads * 64 pairs
        int h = i >> 6, d = i & 63;
        float inv = __powf(10000.0f, -(float)d * (1.0f / 64.0f));
        float s, c; __sincosf(p * inv, &s, &c);
        float x1 = row[4096 + h * 128 + d], x2 = row[4096 + h * 128 + d + 64];
        Kb[(size_t)t * 1024 + h * 128 + d]      = f2bf(x1 * c - x2 * s);
        Kb[(size_t)t * 1024 + h * 128 + d + 64] = f2bf(x2 * c + x1 * s);
    }
    for (int i = tid; i < 1024; i += 256) {      // V transposed: [h][d][t]
        int h = i >> 7, d = i & 127;
        Vt[((size_t)(h * 128 + d)) * 2048 + t] = f2bf(row[5120 + h * 128 + d]);
    }
}

// ---------------------------------------------------------------------------
// Flash attention (causal), bf16 WMMA, fp32 online softmax.
// Grid: (T/64, 32 heads). 4 waves/block, one wave = 16 queries.
// K/V streamed from global (bf16 K+V = 8MB -> L2 resident). Per-wave LDS
// scratch converts P from C-layout to A-layout for the PV WMMAs.
// ---------------------------------------------------------------------------
__global__ __launch_bounds__(128) void attn_fwd(const u16* __restrict__ Q,
                                                const u16* __restrict__ Kt,
                                                const u16* __restrict__ Vt,
                                                u16* __restrict__ Out) {
    __shared__ __align__(16) u16 Pld[4 * 16 * 40];   // per-wave 16x32 P tile, pitch 40

    const int h    = blockIdx.y;
    const int kvh  = h >> 2;
    const int w    = threadIdx.x >> 5;
    const int lane = threadIdx.x & 31;
    const int half = lane >> 4;
    const int ln   = lane & 15;
    const int qb   = blockIdx.x * 64 + w * 16;

    // Q fragments: 4 chunks of K(=d)32, A-layout
    Frag16 qf[4];
    for (int c = 0; c < 4; ++c) {
        const u16* base = Q + (size_t)(qb + ln) * 4096 + h * 128 + c * 32;
        qf[c].u[0] = *(const uint4*)(base + half * 8);
        qf[c].u[1] = *(const uint4*)(base + 16 + half * 8);
    }

    v8f zero = {0.f,0.f,0.f,0.f,0.f,0.f,0.f,0.f};
    v8f o[8];
    float mr[8], lr[8];
    for (int f = 0; f < 8; ++f) o[f] = zero;
    for (int r = 0; r < 8; ++r) { mr[r] = -6.0e29f; lr[r] = 0.0f; }

    u16* pbase = Pld + w * 640;
    const int nkb = (qb + 16 + 31) >> 5;             // causal: key blocks needed

    for (int kbi = 0; kbi < nkb; ++kbi) {
        const int kb = kbi * 32;

        // S = Q * K^T over d=128 (4 WMMAs per 16-key tile)
        v8f s0 = zero, s1 = zero;
        for (int c = 0; c < 4; ++c) {
            Frag16 kf0, kf1;
            const u16* k0 = Kt + (size_t)(kb + ln) * 1024 + kvh * 128 + c * 32;
            kf0.u[0] = *(const uint4*)(k0 + half * 8);
            kf0.u[1] = *(const uint4*)(k0 + 16 + half * 8);
            const u16* k1 = Kt + (size_t)(kb + 16 + ln) * 1024 + kvh * 128 + c * 32;
            kf1.u[0] = *(const uint4*)(k1 + half * 8);
            kf1.u[1] = *(const uint4*)(k1 + 16 + half * 8);
            s0 = WMMA_BF16(qf[c].v, kf0.v, s0);
            s1 = WMMA_BF16(qf[c].v, kf1.v, s1);
        }

        // Online softmax in C-layout: row = r + 8*half, col = ln (+16 for s1)
        const int key0 = kb + ln, key1 = kb + 16 + ln;
        for (int r = 0; r < 8; ++r) {
            const int qrow = qb + r + half * 8;
            float a = (key0 <= qrow) ? s0[r] : -1.0e30f;
            float b = (key1 <= qrow) ? s1[r] : -1.0e30f;
            float mx = fmaxf(a, b);
            mx = fmaxf(mx, __shfl_xor(mx, 8, 32));
            mx = fmaxf(mx, __shfl_xor(mx, 4, 32));
            mx = fmaxf(mx, __shfl_xor(mx, 2, 32));
            mx = fmaxf(mx, __shfl_xor(mx, 1, 32));
            float mnew = fmaxf(mr[r], mx);
            float corr = __expf(mr[r] - mnew);
            mr[r] = mnew;
            float p0 = __expf(a - mnew);
            float p1 = __expf(b - mnew);
            float rs = p0 + p1;
            rs += __shfl_xor(rs, 8, 32);
            rs += __shfl_xor(rs, 4, 32);
            rs += __shfl_xor(rs, 2, 32);
            rs += __shfl_xor(rs, 1, 32);
            lr[r] = lr[r] * corr + rs;
            for (int f = 0; f < 8; ++f) o[f][r] *= corr;
            const int prow = r + half * 8;
            pbase[prow * 40 + ln]      = f2bf(p0);
            pbase[prow * 40 + 16 + ln] = f2bf(p1);
        }

        // wave-private LDS round trip: C-layout -> A-layout
        asm volatile("s_wait_dscnt 0x0" ::: "memory");
        Frag16 pf;
        pf.u[0] = *(const uint4*)(pbase + ln * 40 + half * 8);
        pf.u[1] = *(const uint4*)(pbase + ln * 40 + 16 + half * 8);

        // O += P * V  (V transposed [d][t]: contiguous key runs per lane)
        for (int f = 0; f < 8; ++f) {
            Frag16 vf;
            const u16* vb = Vt + (size_t)(kvh * 128 + f * 16 + ln) * 2048 + kb;
            vf.u[0] = *(const uint4*)(vb + half * 8);
            vf.u[1] = *(const uint4*)(vb + 16 + half * 8);
            o[f] = WMMA_BF16(pf.v, vf.v, o[f]);
        }
    }

    for (int f = 0; f < 8; ++f)
        for (int r = 0; r < 8; ++r) {
            const int t = qb + r + half * 8;
            Out[(size_t)t * 4096 + h * 128 + f * 16 + ln] = f2bf(o[f][r] / lr[r]);
        }
}

// ---------------------------------------------------------------------------
extern "C" void kernel_launch(void* const* d_in, const int* in_sizes, int n_in,
                              void* d_out, int out_size, void* d_ws, size_t ws_size,
                              hipStream_t stream) {
    const float* hs   = (const float*)d_in[0];   // [2048, 4096]
    const int*   pos  = (const int*)  d_in[1];   // [2048]
    const float* wqkv = (const float*)d_in[2];   // [6144, 4096]
    const float* wo   = (const float*)d_in[3];   // [4096, 4096]
    float* out = (float*)d_out;                  // [2048, 4096]

    const long nHS   = 2048L * 4096;
    const long nWQKV = 6144L * 4096;
    const long nWO   = 4096L * 4096;

    u16* hs_b   = (u16*)d_ws;
    u16* wqkv_b = hs_b + nHS;
    u16* wo_b   = wqkv_b + nWQKV;
    float* qkv  = (float*)(wo_b + nWO);          // [2048, 6144] f32
    u16* qB     = (u16*)(qkv + 2048L * 6144);    // [2048, 32, 128]
    u16* kB     = qB + 2048L * 4096;             // [2048, 8, 128]
    u16* vT     = kB + 2048L * 1024;             // [8, 128, 2048]
    u16* attnB  = vT + 2048L * 1024;             // [2048, 4096]

    cvt_f32_bf16<<<(int)(nHS   / 2048), 256, 0, stream>>>(hs,   hs_b,   nHS);
    cvt_f32_bf16<<<(int)(nWQKV / 2048), 256, 0, stream>>>(wqkv, wqkv_b, nWQKV);
    cvt_f32_bf16<<<(int)(nWO   / 2048), 256, 0, stream>>>(wo,   wo_b,   nWO);

    gemm_bf16_tn<<<dim3(6144 / 128, 2048 / 128), 256, 0, stream>>>(
        hs_b, wqkv_b, qkv, 2048, 6144, 4096);

    rope_pack<<<2048, 256, 0, stream>>>(qkv, pos, qB, kB, vT);

    attn_fwd<<<dim3(2048 / 64, 32), 128, 0, stream>>>(qB, kB, vT, attnB);

    gemm_bf16_tn<<<dim3(4096 / 128, 2048 / 128), 256, 0, stream>>>(
        attnB, wo_b, out, 2048, 4096, 4096);
}
